// ScoreNet_82386062672000
// MI455X (gfx1250) — matrix operand order repaired and checked by reading, not compiled, not run on hardware
//
#include <hip/hip_runtime.h>

typedef __attribute__((ext_vector_type(16))) __bf16 v16bf;
typedef __attribute__((ext_vector_type(8)))  __bf16 v8bf;
typedef __attribute__((ext_vector_type(8)))  float  v8f;
typedef __attribute__((address_space(3)))    __bf16 lds_bf16;

#define NBATCH 8
#define NSEQ   2048
#define CDIM   256
#define KP     224                 // padded P/V inner width (192 + 16 + 16 pad)
#define NKSTEP (KP / 32)           // 7 K-chunks of 32
#define NCOLS  448                 // combined projection output columns (2*KP)
#define HSCALE 0.17677669529663687f // 1/sqrt(32)

static __device__ __forceinline__ __bf16 f2bf(float f) {
    unsigned u = __float_as_uint(f);
    unsigned r = (u + 0x7FFFu + ((u >> 16) & 1u)) >> 16;   // RNE
    unsigned short s = (unsigned short)r;
    return __builtin_bit_cast(__bf16, s);
}
static __device__ __forceinline__ float bf2f(__bf16 h) {
    unsigned short s = __builtin_bit_cast(unsigned short, h);
    return __uint_as_float(((unsigned)s) << 16);
}
static __device__ __forceinline__ v16bf cmb(v8bf a, v8bf b) {
    return __builtin_shufflevector(a, b, 0,1,2,3,4,5,6,7,8,9,10,11,12,13,14,15);
}
static __device__ __forceinline__ v8f wmma_bf16(v16bf a, v16bf b, v8f c) {
    // (neg_a, A, neg_b, B, c_mod, C, reuse_a, reuse_b)
    return __builtin_amdgcn_wmma_f32_16x16x32_bf16(false, a, false, b, (short)0, c, false, false);
}
// LDS byte address (offset from LDS base) of a pointer into a __shared__ array
static __device__ __forceinline__ unsigned lds_addr(const __bf16* p) {
    return (unsigned)reinterpret_cast<unsigned long long>((lds_bf16*)p);
}
// Direct global -> LDS async copy of 32 bytes (two b128 beats); ASYNCcnt tracked.
// INST_OFFSET is added to both LDS and global addresses (ISA 08_async_tensor §4.4),
// and our LDS chunk layout mirrors the global layout, so offset:16 works for both.
static __device__ __forceinline__ void async_copy32(unsigned lds_byte_addr, const __bf16* gptr) {
    asm volatile("global_load_async_to_lds_b128 %0, %1, off"
                 :: "v"(lds_byte_addr), "v"(gptr) : "memory");
    asm volatile("global_load_async_to_lds_b128 %0, %1, off offset:16"
                 :: "v"(lds_byte_addr), "v"(gptr) : "memory");
}

// ---------------------------------------------------------------------------
// Kernel 1: fold adjacent vertices, split x into bf16 hi/lo
// ---------------------------------------------------------------------------
__global__ void k_fold(const float* __restrict__ feats,
                       __bf16* __restrict__ xhi, __bf16* __restrict__ xlo) {
    size_t g = (size_t)blockIdx.x * 256 + threadIdx.x;   // < 8*2048*256
    int c = (int)(g & 255);
    size_t row = g >> 8;                 // b*2048 + n
    int b = (int)(row >> 11), n = (int)(row & 2047);
    size_t base = ((size_t)b * 4097 + 1 + 2 * (size_t)n) * 256 + c;
    float x = 0.5f * (feats[base] + feats[base + 256]);
    __bf16 h = f2bf(x);
    xhi[g] = h;
    xlo[g] = f2bf(x - bf2f(h));
}

// ---------------------------------------------------------------------------
// Kernel 2: build combined transposed weight Wt[j][c], j in [0,448), hi/lo split
//   j<192: Wq  | 192..207: Wpi | 208..223: 0 | 224..415: Wk | 416..431: Wpj | pad 0
// ---------------------------------------------------------------------------
__global__ void k_wsplit(const float* __restrict__ Wq, const float* __restrict__ Wk,
                         const float* __restrict__ Wpi, const float* __restrict__ Wpj,
                         __bf16* __restrict__ wthi, __bf16* __restrict__ wtlo) {
    int g = blockIdx.x * 256 + threadIdx.x;  // < 448*256
    int c = g & 255;
    int j = g >> 8;
    float w;
    if (j < 192)       w = Wq[c * 192 + j];
    else if (j < 208)  w = Wpi[c * 16 + (j - 192)];
    else if (j < 224)  w = 0.f;
    else if (j < 416)  w = Wk[c * 192 + (j - 224)];
    else if (j < 432)  w = Wpj[c * 16 + (j - 416)];
    else               w = 0.f;
    size_t idx = (size_t)j * 256 + c;        // transposed: K contiguous per column
    __bf16 h = f2bf(w);
    wthi[idx] = h;
    wtlo[idx] = f2bf(w - bf2f(h));
}

// ---------------------------------------------------------------------------
// Kernel 3: projection GEMM  Y(16384 x 448) = x(16384 x 256) @ Wc(256 x 448)
// bf16x3 WMMA; epilogue applies scales/relu and re-splits P,V to bf16 hi/lo
// grid: (448/64, 16384/128), 256 threads (8 waves; wave = 16 rows x 64 cols)
// ---------------------------------------------------------------------------
__global__ __launch_bounds__(256) void k_proj(
    const __bf16* __restrict__ xhi, const __bf16* __restrict__ xlo,
    const __bf16* __restrict__ wthi, const __bf16* __restrict__ wtlo,
    const float* __restrict__ bpi, const float* __restrict__ bpj,
    const float* __restrict__ alphap, const float* __restrict__ lsp,
    __bf16* __restrict__ Phi, __bf16* __restrict__ Plo,
    __bf16* __restrict__ Vhi, __bf16* __restrict__ Vlo) {
    int tid  = threadIdx.x;
    int lane = tid & 31, w = tid >> 5;
    int tileN = blockIdx.x * 64;
    int Mbase = blockIdx.y * 128 + w * 16;

    v8f acc[4] = {};
    int arow = Mbase + (lane & 15);
    int asel = (lane < 16) ? 0 : 16;    // A layout: lanes<16 K 0..15, else K 16..31
    int bsel = (lane < 16) ? 0 : 16;    // B layout: lanes<16 K 0..15, else K 16..31

    for (int k0 = 0; k0 < CDIM; k0 += 32) {
        const __bf16* ah_p = xhi + (size_t)arow * CDIM + k0 + asel;
        const __bf16* al_p = xlo + (size_t)arow * CDIM + k0 + asel;
        v16bf ah = cmb(*(const v8bf*)ah_p, *(const v8bf*)(ah_p + 8));
        v16bf al = cmb(*(const v8bf*)al_p, *(const v8bf*)(al_p + 8));
#pragma unroll
        for (int s = 0; s < 4; ++s) {
            int ncol = tileN + s * 16 + (lane & 15);
            const __bf16* bh_p = wthi + (size_t)ncol * CDIM + k0 + bsel;
            const __bf16* bl_p = wtlo + (size_t)ncol * CDIM + k0 + bsel;
            v16bf bh = cmb(*(const v8bf*)bh_p, *(const v8bf*)(bh_p + 8));
            v16bf bl = cmb(*(const v8bf*)bl_p, *(const v8bf*)(bl_p + 8));
            acc[s] = wmma_bf16(ah, bh, acc[s]);
            acc[s] = wmma_bf16(ah, bl, acc[s]);
            acc[s] = wmma_bf16(al, bh, acc[s]);
        }
    }

    float ls = fmaxf(lsp[0], 0.01f);
    float ap = fmaxf(alphap[0], 0.f) * ls;
    float qs = ls * HSCALE;
    int rbase = Mbase + ((lane >= 16) ? 8 : 0);
#pragma unroll
    for (int s = 0; s < 4; ++s) {
        int j = tileN + s * 16 + (lane & 15);
#pragma unroll
        for (int r = 0; r < 8; ++r) {
            size_t g = (size_t)(rbase + r);      // flattened b*2048+n
            float y = acc[s][r];
            float o;
            __bf16 *dh, *dl;
            size_t idx;
            if (j < KP) {
                if (j < 192)      o = y * qs;
                else if (j < 208) o = ap * fmaxf(y + bpi[j - 192], 0.f);
                else              o = 0.f;
                idx = g * KP + j; dh = Phi; dl = Plo;
            } else {
                int jj = j - KP;
                if (jj < 192)      o = y;
                else if (jj < 208) o = fmaxf(y + bpj[jj - 192], 0.f);
                else               o = 0.f;
                idx = g * KP + jj; dh = Vhi; dl = Vlo;
            }
            __bf16 h = f2bf(o);
            dh[idx] = h;
            dl[idx] = f2bf(o - bf2f(h));
        }
    }
}

// ---------------------------------------------------------------------------
// Kernel 4: diagonal vector  d[g] = ls * (x[g]·Wd + bd)
// ---------------------------------------------------------------------------
__global__ void k_dvec(const float* __restrict__ feats, const float* __restrict__ Wd,
                       const float* __restrict__ bd, const float* __restrict__ lsp,
                       float* __restrict__ dd) {
    int g = blockIdx.x * 256 + threadIdx.x;   // < 16384
    int b = g >> 11, n = g & 2047;
    const float* r0 = feats + ((size_t)b * 4097 + 1 + 2 * (size_t)n) * 256;
    float acc = 0.f;
#pragma unroll 4
    for (int c = 0; c < 256; ++c)
        acc += 0.5f * (r0[c] + r0[c + 256]) * Wd[c];
    dd[g] = fmaxf(lsp[0], 0.01f) * (acc + bd[0]);
}

// ---------------------------------------------------------------------------
// Kernel 5: score GEMM per batch: S = P · Vᵀ (+ diag), 128x128 tile per block
// grid: (16, 16, 8), 256 threads = 8 waves
// V K-chunks double-buffered in LDS via GLOBAL_LOAD_ASYNC_TO_LDS_B128:
// chunk k+1's copy overlaps chunk k's WMMAs; s_wait_asynccnt 4 retires the
// older chunk (async loads complete in order, 4 ops in flight per chunk).
// ---------------------------------------------------------------------------
__global__ __launch_bounds__(256) void k_score(
    const __bf16* __restrict__ Phi, const __bf16* __restrict__ Plo,
    const __bf16* __restrict__ Vhi, const __bf16* __restrict__ Vlo,
    const float* __restrict__ dd, float* __restrict__ out) {
    __shared__ __bf16 sBh[2][128 * 32];
    __shared__ __bf16 sBl[2][128 * 32];

    int tid = threadIdx.x, lane = tid & 31, w = tid >> 5;
    int b = blockIdx.z;
    int tileN = blockIdx.x * 128;
    int tileM = blockIdx.y * 128;
    size_t pb = (size_t)b * NSEQ * KP;

    v8f acc[8] = {};

    int arow = tileM + w * 16 + (lane & 15);
    int asel = (lane < 16) ? 0 : 16;
    const __bf16* Pah = Phi + pb + (size_t)arow * KP;
    const __bf16* Pal = Plo + pb + (size_t)arow * KP;

    int ldr = tid >> 1;                 // 0..127: V row within tile
    int ldk = (tid & 1) * 16;           // which K half of the 32-chunk
    const __bf16* Vrh = Vhi + pb + (size_t)(tileN + ldr) * KP + ldk;
    const __bf16* Vrl = Vlo + pb + (size_t)(tileN + ldr) * KP + ldk;
    unsigned ldsH[2] = { lds_addr(&sBh[0][ldr * 32 + ldk]), lds_addr(&sBh[1][ldr * 32 + ldk]) };
    unsigned ldsL[2] = { lds_addr(&sBl[0][ldr * 32 + ldk]), lds_addr(&sBl[1][ldr * 32 + ldk]) };

    int bk = (lane < 16) ? 0 : 16;

    // prologue: issue chunk 0 into buffer 0 (4 async b128 ops per thread)
    async_copy32(ldsH[0], Vrh);
    async_copy32(ldsL[0], Vrl);

    for (int ks = 0; ks < NKSTEP; ++ks) {
        int cur = ks & 1;
        if (ks + 1 < NKSTEP) {
            // issue next chunk into the other buffer, then retire the older 4 ops
            async_copy32(ldsH[cur ^ 1], Vrh + (ks + 1) * 32);
            async_copy32(ldsL[cur ^ 1], Vrl + (ks + 1) * 32);
            asm volatile("s_wait_asynccnt 0x4" ::: "memory");
        } else {
            asm volatile("s_wait_asynccnt 0x0" ::: "memory");
        }
        __syncthreads();                // all waves' chunk-ks data visible in LDS

        int k0 = ks * 32;
        const __bf16* pah = Pah + k0 + asel;
        const __bf16* pal = Pal + k0 + asel;
        v16bf ah = cmb(*(const v8bf*)pah, *(const v8bf*)(pah + 8));
        v16bf al = cmb(*(const v8bf*)pal, *(const v8bf*)(pal + 8));
#pragma unroll
        for (int s = 0; s < 8; ++s) {
            const __bf16* ph = &sBh[cur][(s * 16 + (lane & 15)) * 32 + bk];
            const __bf16* pl = &sBl[cur][(s * 16 + (lane & 15)) * 32 + bk];
            v16bf bh = cmb(*(const v8bf*)ph, *(const v8bf*)(ph + 8));
            v16bf bl = cmb(*(const v8bf*)pl, *(const v8bf*)(pl + 8));
            acc[s] = wmma_bf16(ah, bh, acc[s]);
            acc[s] = wmma_bf16(ah, bl, acc[s]);
            acc[s] = wmma_bf16(al, bh, acc[s]);
        }
        __syncthreads();                // all waves done reading buf[cur] before reissue
    }

    int rbase = tileM + w * 16 + ((lane >= 16) ? 8 : 0);
    const float* ddb = dd + b * NSEQ;
    float* ob = out + (size_t)b * NSEQ * NSEQ;
#pragma unroll
    for (int s = 0; s < 8; ++s) {
        int col = tileN + s * 16 + (lane & 15);
#pragma unroll
        for (int r = 0; r < 8; ++r) {
            int row = rbase + r;
            float v = acc[s][r];
            if (row == col) v += ddb[row];
            ob[(size_t)row * NSEQ + col] = v;
        }
    }
}

// ---------------------------------------------------------------------------
extern "C" void kernel_launch(void* const* d_in, const int* in_sizes, int n_in,
                              void* d_out, int out_size, void* d_ws, size_t ws_size,
                              hipStream_t stream) {
    (void)in_sizes; (void)n_in; (void)out_size; (void)ws_size;
    const float* feats = (const float*)d_in[0];
    const float* Wq    = (const float*)d_in[1];
    const float* Wk    = (const float*)d_in[2];
    const float* Wpi   = (const float*)d_in[3];
    const float* bpi   = (const float*)d_in[4];
    const float* Wpj   = (const float*)d_in[5];
    const float* bpj   = (const float*)d_in[6];
    const float* Wd    = (const float*)d_in[7];
    const float* bd    = (const float*)d_in[8];
    const float* alpha = (const float*)d_in[9];
    const float* lsp   = (const float*)d_in[10];
    float* out = (float*)d_out;

    char* ws = (char*)d_ws;
    size_t off = 0;
    auto carve = [&](size_t bytes) -> char* {
        char* p = ws + off;
        off += (bytes + 255) & ~(size_t)255;
        return p;
    };
    const size_t XN  = (size_t)NBATCH * NSEQ * CDIM;   // 4,194,304
    const size_t WTN = (size_t)NCOLS * CDIM;           // 114,688
    const size_t PVN = (size_t)NBATCH * NSEQ * KP;     // 3,670,016
    const size_t DN  = (size_t)NBATCH * NSEQ;          // 16,384

    __bf16* xhi  = (__bf16*)carve(XN * 2);
    __bf16* xlo  = (__bf16*)carve(XN * 2);
    __bf16* wthi = (__bf16*)carve(WTN * 2);
    __bf16* wtlo = (__bf16*)carve(WTN * 2);
    __bf16* Phi  = (__bf16*)carve(PVN * 2);
    __bf16* Plo  = (__bf16*)carve(PVN * 2);
    __bf16* Vhi  = (__bf16*)carve(PVN * 2);
    __bf16* Vlo  = (__bf16*)carve(PVN * 2);
    float*  dd   = (float*)carve(DN * 4);

    k_fold<<<dim3((unsigned)(XN / 256)), dim3(256), 0, stream>>>(feats, xhi, xlo);
    k_wsplit<<<dim3((unsigned)(WTN / 256)), dim3(256), 0, stream>>>(Wq, Wk, Wpi, Wpj, wthi, wtlo);
    k_proj<<<dim3(NCOLS / 64, (NBATCH * NSEQ) / 128), dim3(256), 0, stream>>>(
        xhi, xlo, wthi, wtlo, bpi, bpj, alpha, lsp, Phi, Plo, Vhi, Vlo);
    k_dvec<<<dim3((unsigned)(DN / 256)), dim3(256), 0, stream>>>(feats, Wd, bd, lsp, dd);
    k_score<<<dim3(NSEQ / 128, NSEQ / 128, NBATCH), dim3(256), 0, stream>>>(
        Phi, Plo, Vhi, Vlo, dd, out);
}
